// UdopAttention_197568495659
// MI455X (gfx1250) — compile-verified, matrix operand order checked by reading, not run on hardware
//
#include <hip/hip_runtime.h>

#define S_LEN   2048
#define D_MODEL 1024
#define H_NUM   16
#define DKV     64
#define BATCH   2

typedef __attribute__((ext_vector_type(16))) __bf16 v16bf;
typedef __attribute__((ext_vector_type(8)))  float  v8f;

union Frag16 {
    v16bf v;
    unsigned int u[8];
};

__device__ __forceinline__ unsigned short f2bf(float f) {
    unsigned int u = __builtin_bit_cast(unsigned int, f);
    u += 0x7FFFu + ((u >> 16) & 1u);     // round-to-nearest-even
    return (unsigned short)(u >> 16);
}

__device__ __forceinline__ unsigned int f2bf2(float lo, float hi) {
    return (unsigned int)f2bf(lo) | ((unsigned int)f2bf(hi) << 16);
}

// K index for fragment dword v, given lane's kbase (0 or 8).
// Matches ISA 16-bit A/B layout: VGPR0..3 -> K={kb..kb+7}, VGPR4..7 -> K=16+{..}
__device__ __forceinline__ int kmap(int v, int kb) {
    return ((v & 4) ? 16 : 0) + kb + 2 * (v & 3);
}

// ---------------------------------------------------------------------------
// Kernel 0: precompute T5 relative-position bias table  [H][4096] (4095 used, idx = rel+2047)
// ---------------------------------------------------------------------------
__global__ void bias_table_kernel(const float* __restrict__ rel_bias,
                                  float* __restrict__ table) {
    int tid = blockIdx.x * blockDim.x + threadIdx.x;
    if (tid >= H_NUM * 4095) return;
    int h   = tid / 4095;
    int idx = tid % 4095;
    int rel = idx - 2047;
    int b   = (rel > 0) ? 16 : 0;          // num_buckets/2
    int rp  = (rel < 0) ? -rel : rel;
    int bucket;
    if (rp < 8) {                           // max_exact = 8
        bucket = b + rp;
    } else {
        // 8 + log(rp/8)/log(128/8) * 8 ; truncate like astype(int32)
        int large = 8 + (int)(__logf((float)rp * 0.125f) * (8.0f / 2.772588722f));
        if (large > 15) large = 15;
        bucket = b + large;
    }
    table[h * 4096 + idx] = rel_bias[bucket * H_NUM + h];
}

// ---------------------------------------------------------------------------
// Kernel 1: fused QKV projection.  hidden[4096,1024] x W[1024,1024] (z picks W)
// Output bf16 in [B,H,S,Dk] layout for the attention kernel.
// ---------------------------------------------------------------------------
__global__ __launch_bounds__(128)
void qkv_gemm_kernel(const float* __restrict__ hidden,
                     const float* __restrict__ Wq,
                     const float* __restrict__ Wk,
                     const float* __restrict__ Wv,
                     unsigned short* __restrict__ qw,
                     unsigned short* __restrict__ kw,
                     unsigned short* __restrict__ vw) {
    __shared__ __align__(16) unsigned short a_lds[64 * 36];   // A tile, row-major
    __shared__ __align__(16) unsigned short bt_lds[64 * 36];  // B tile, transposed [n][k]

    const int t    = threadIdx.x;
    const int wave = t >> 5;
    const int lane = t & 31;
    const int ntile = blockIdx.x * 64;
    const int mtile = blockIdx.y * 64;
    const int which = blockIdx.z;
    const float* W = (which == 0) ? Wq : (which == 1) ? Wk : Wv;
    unsigned short* dst = (which == 0) ? qw : (which == 1) ? kw : vw;

    const int m_lane   = lane & 15;
    const int kb       = (lane < 16) ? 0 : 8;
    const int half_off = (lane < 16) ? 0 : 8;

    v8f acc[4] = {};

    for (int k0 = 0; k0 < D_MODEL; k0 += 32) {
        // A tile 64x32 (fp32 -> bf16), float4 loads, b64 LDS stores
        #pragma unroll
        for (int i = 0; i < 4; ++i) {
            int e = i * 128 + t;
            int row = e >> 3, kg = (e & 7) * 4;
            float4 f = *(const float4*)&hidden[(size_t)(mtile + row) * D_MODEL + k0 + kg];
            uint2 p;
            p.x = f2bf2(f.x, f.y);
            p.y = f2bf2(f.z, f.w);
            *(uint2*)&a_lds[row * 36 + kg] = p;
        }
        // B tile 32x64, float4 loads, stored transposed (scatter b16)
        #pragma unroll
        for (int i = 0; i < 4; ++i) {
            int e = i * 128 + t;
            int k = e >> 4, ng = (e & 15) * 4;
            float4 f = *(const float4*)&W[(size_t)(k0 + k) * D_MODEL + ntile + ng];
            bt_lds[(ng + 0) * 36 + k] = f2bf(f.x);
            bt_lds[(ng + 1) * 36 + k] = f2bf(f.y);
            bt_lds[(ng + 2) * 36 + k] = f2bf(f.z);
            bt_lds[(ng + 3) * 36 + k] = f2bf(f.w);
        }
        if (k0 + 32 < D_MODEL) {
            __builtin_prefetch(&hidden[(size_t)(mtile + (t & 63)) * D_MODEL + k0 + 32], 0, 3);
            __builtin_prefetch(&W[(size_t)(k0 + 32 + (t >> 2)) * D_MODEL + ntile], 0, 3);
        }
        __syncthreads();

        Frag16 af;
        const int arow = wave * 16 + m_lane;
        #pragma unroll
        for (int v = 0; v < 8; ++v)
            af.u[v] = *(const unsigned int*)&a_lds[arow * 36 + kmap(v, kb)];

        #pragma unroll
        for (int a = 0; a < 4; ++a) {
            Frag16 bfb;
            int ncol = a * 16 + m_lane;
            #pragma unroll
            for (int v = 0; v < 8; ++v)
                bfb.u[v] = *(const unsigned int*)&bt_lds[ncol * 36 + kmap(v, kb)];
            acc[a] = __builtin_amdgcn_wmma_f32_16x16x32_bf16(
                false, af.v, false, bfb.v, (short)0, acc[a], false, false);
        }
        __syncthreads();
    }

    // scatter into [B,H,S,Dk] bf16
    #pragma unroll
    for (int a = 0; a < 4; ++a) {
        int ncol = ntile + a * 16 + m_lane;
        int h = ncol >> 6, d = ncol & 63;
        #pragma unroll
        for (int r = 0; r < 8; ++r) {
            int m = mtile + wave * 16 + r + half_off;
            int b = m >> 11, s = m & 2047;
            dst[(((size_t)(b * H_NUM + h) * S_LEN) + s) * DKV + d] = f2bf(acc[a][r]);
        }
    }
}

// ---------------------------------------------------------------------------
// Kernel 2: flash attention per (b,h).  64 Q rows per workgroup, 4 waves x 16.
// ---------------------------------------------------------------------------
__global__ __launch_bounds__(128)
void attn_kernel(const unsigned short* __restrict__ qw,
                 const unsigned short* __restrict__ kw,
                 const unsigned short* __restrict__ vw,
                 const float* __restrict__ bias_table,
                 unsigned short* __restrict__ att) {
    __shared__ __align__(16) unsigned short kt[64 * 72];        // K tile row-major [n][d]
    __shared__ __align__(16) unsigned short vt[64 * 72];        // V tile transposed [d][k]
    __shared__ __align__(16) unsigned short pt[4 * 16 * 72];    // per-wave P strips

    const int t    = threadIdx.x;
    const int wave = t >> 5;
    const int lane = t & 31;
    const int qtile = blockIdx.x * 64;
    const int bh    = blockIdx.y;
    const int h     = bh & (H_NUM - 1);

    const int m_lane   = lane & 15;
    const int kb       = (lane < 16) ? 0 : 8;
    const int half_off = (lane < 16) ? 0 : 8;

    const size_t base = (size_t)bh * S_LEN * DKV;

    // Q fragments (held for the whole pass): row = qtile + wave*16 + m_lane
    Frag16 qf[2];
    {
        const unsigned short* qrow = qw + base + (size_t)(qtile + wave * 16 + m_lane) * DKV;
        #pragma unroll
        for (int c = 0; c < 2; ++c)
            #pragma unroll
            for (int v = 0; v < 8; ++v)
                qf[c].u[v] = *(const unsigned int*)&qrow[c * 32 + kmap(v, kb)];
    }

    float run_max[8], run_sum[8];
    #pragma unroll
    for (int r = 0; r < 8; ++r) { run_max[r] = -1e30f; run_sum[r] = 0.0f; }
    v8f o_acc[4] = {};

    unsigned short* pw = &pt[wave * 16 * 72];

    for (int kt0 = 0; kt0 < S_LEN; kt0 += 64) {
        __syncthreads();
        // cooperative load: K row-major (b128), V transposed (b16 scatter)
        #pragma unroll
        for (int i = 0; i < 4; ++i) {
            int e   = i * 128 + t;
            int row = e >> 3;
            int dg  = (e & 7) * 8;
            uint4 kq = *(const uint4*)&kw[base + (size_t)(kt0 + row) * DKV + dg];
            *(uint4*)&kt[row * 72 + dg] = kq;
            uint4 vq = *(const uint4*)&vw[base + (size_t)(kt0 + row) * DKV + dg];
            vt[(dg + 0) * 72 + row] = (unsigned short)(vq.x & 0xFFFFu);
            vt[(dg + 1) * 72 + row] = (unsigned short)(vq.x >> 16);
            vt[(dg + 2) * 72 + row] = (unsigned short)(vq.y & 0xFFFFu);
            vt[(dg + 3) * 72 + row] = (unsigned short)(vq.y >> 16);
            vt[(dg + 4) * 72 + row] = (unsigned short)(vq.z & 0xFFFFu);
            vt[(dg + 5) * 72 + row] = (unsigned short)(vq.z >> 16);
            vt[(dg + 6) * 72 + row] = (unsigned short)(vq.w & 0xFFFFu);
            vt[(dg + 7) * 72 + row] = (unsigned short)(vq.w >> 16);
        }
        if (kt0 + 64 < S_LEN) {
            __builtin_prefetch(&kw[base + (size_t)(kt0 + 64 + (t >> 1)) * DKV], 0, 3);
            __builtin_prefetch(&vw[base + (size_t)(kt0 + 64 + (t >> 1)) * DKV], 0, 3);
        }
        __syncthreads();

        // scores S = Q K^T  (4 n-subtiles, K over Dk=64 in 2 chunks)
        v8f sacc[4] = {};
        #pragma unroll
        for (int c = 0; c < 2; ++c) {
            #pragma unroll
            for (int a = 0; a < 4; ++a) {
                Frag16 bfb;
                int n = a * 16 + m_lane;
                #pragma unroll
                for (int v = 0; v < 8; ++v)
                    bfb.u[v] = *(const unsigned int*)&kt[n * 72 + c * 32 + kmap(v, kb)];
                sacc[a] = __builtin_amdgcn_wmma_f32_16x16x32_bf16(
                    false, qf[c].v, false, bfb.v, (short)0, sacc[a], false, false);
            }
        }

        // add relative-position bias (C layout: row = r+half_off, col = lane&15)
        #pragma unroll
        for (int a = 0; a < 4; ++a) {
            int kpos = kt0 + a * 16 + m_lane;
            #pragma unroll
            for (int r = 0; r < 8; ++r) {
                int qpos = qtile + wave * 16 + r + half_off;
                sacc[a][r] += bias_table[h * 4096 + (kpos - qpos + 2047)];
            }
        }

        // online softmax (rows reduced across the 16 lanes of each half)
        float corr[8];
        #pragma unroll
        for (int r = 0; r < 8; ++r) {
            float tmax = fmaxf(fmaxf(sacc[0][r], sacc[1][r]), fmaxf(sacc[2][r], sacc[3][r]));
            #pragma unroll
            for (int msk = 1; msk < 16; msk <<= 1)
                tmax = fmaxf(tmax, __shfl_xor(tmax, msk, 32));
            float mnew = fmaxf(run_max[r], tmax);
            corr[r] = __expf(run_max[r] - mnew);
            run_max[r] = mnew;
            float ts = 0.0f;
            #pragma unroll
            for (int a = 0; a < 4; ++a) {
                float p = __expf(sacc[a][r] - mnew);
                sacc[a][r] = p;
                ts += p;
            }
            #pragma unroll
            for (int msk = 1; msk < 16; msk <<= 1)
                ts += __shfl_xor(ts, msk, 32);
            run_sum[r] = run_sum[r] * corr[r] + ts;
        }
        #pragma unroll
        for (int a = 0; a < 4; ++a)
            #pragma unroll
            for (int r = 0; r < 8; ++r)
                o_acc[a][r] *= corr[r];

        // P: C layout -> LDS -> A layout
        #pragma unroll
        for (int a = 0; a < 4; ++a)
            #pragma unroll
            for (int r = 0; r < 8; ++r)
                pw[(r + half_off) * 72 + a * 16 + m_lane] = f2bf(sacc[a][r]);
        __syncthreads();

        // O += P @ V   (K over tile = 64, 2 chunks)
        #pragma unroll
        for (int c = 0; c < 2; ++c) {
            Frag16 pf;
            #pragma unroll
            for (int v = 0; v < 8; ++v)
                pf.u[v] = *(const unsigned int*)&pw[m_lane * 72 + c * 32 + kmap(v, kb)];
            #pragma unroll
            for (int a = 0; a < 4; ++a) {
                Frag16 bfb;
                int d = a * 16 + m_lane;
                #pragma unroll
                for (int v = 0; v < 8; ++v)
                    bfb.u[v] = *(const unsigned int*)&vt[d * 72 + c * 32 + kmap(v, kb)];
                o_acc[a] = __builtin_amdgcn_wmma_f32_16x16x32_bf16(
                    false, pf.v, false, bfb.v, (short)0, o_acc[a], false, false);
            }
        }
    }

    // normalize and write attn output as bf16 [B,S,H*Dk]
    #pragma unroll
    for (int r = 0; r < 8; ++r) {
        float inv = 1.0f / run_sum[r];
        int qpos = qtile + wave * 16 + r + half_off;
        int b    = bh >> 4;
        size_t rowbase = ((size_t)(b * S_LEN + qpos) * (H_NUM * DKV)) + h * DKV;
        #pragma unroll
        for (int a = 0; a < 4; ++a)
            att[rowbase + a * 16 + m_lane] = f2bf(o_acc[a][r] * inv);
    }
}

// ---------------------------------------------------------------------------
// Kernel 3: output projection  att(bf16)[4096,1024] x Wo(f32)[1024,1024] -> f32
// ---------------------------------------------------------------------------
__global__ __launch_bounds__(128)
void out_gemm_kernel(const unsigned short* __restrict__ att,
                     const float* __restrict__ Wo,
                     float* __restrict__ out) {
    __shared__ __align__(16) unsigned short a_lds[64 * 36];
    __shared__ __align__(16) unsigned short bt_lds[64 * 36];

    const int t    = threadIdx.x;
    const int wave = t >> 5;
    const int lane = t & 31;
    const int ntile = blockIdx.x * 64;
    const int mtile = blockIdx.y * 64;

    const int m_lane   = lane & 15;
    const int kb       = (lane < 16) ? 0 : 8;
    const int half_off = (lane < 16) ? 0 : 8;

    v8f acc[4] = {};

    for (int k0 = 0; k0 < D_MODEL; k0 += 32) {
        // A tile 64x32 bf16: b128 loads, b64 LDS stores
        #pragma unroll
        for (int i = 0; i < 2; ++i) {
            int e = i * 128 + t;
            int row = e >> 2, kg = (e & 3) * 8;
            uint4 q = *(const uint4*)&att[(size_t)(mtile + row) * D_MODEL + k0 + kg];
            uint2 lo; lo.x = q.x; lo.y = q.y;
            uint2 hi; hi.x = q.z; hi.y = q.w;
            *(uint2*)&a_lds[row * 36 + kg]     = lo;
            *(uint2*)&a_lds[row * 36 + kg + 4] = hi;
        }
        // B tile 32x64 fp32 -> bf16, stored transposed
        #pragma unroll
        for (int i = 0; i < 4; ++i) {
            int e = i * 128 + t;
            int k = e >> 4, ng = (e & 15) * 4;
            float4 f = *(const float4*)&Wo[(size_t)(k0 + k) * D_MODEL + ntile + ng];
            bt_lds[(ng + 0) * 36 + k] = f2bf(f.x);
            bt_lds[(ng + 1) * 36 + k] = f2bf(f.y);
            bt_lds[(ng + 2) * 36 + k] = f2bf(f.z);
            bt_lds[(ng + 3) * 36 + k] = f2bf(f.w);
        }
        if (k0 + 32 < D_MODEL) {
            __builtin_prefetch(&att[(size_t)(mtile + (t & 63)) * D_MODEL + k0 + 32], 0, 3);
            __builtin_prefetch(&Wo[(size_t)(k0 + 32 + (t >> 2)) * D_MODEL + ntile], 0, 3);
        }
        __syncthreads();

        Frag16 af;
        const int arow = wave * 16 + m_lane;
        #pragma unroll
        for (int v = 0; v < 8; ++v)
            af.u[v] = *(const unsigned int*)&a_lds[arow * 36 + kmap(v, kb)];

        #pragma unroll
        for (int a = 0; a < 4; ++a) {
            Frag16 bfb;
            int ncol = a * 16 + m_lane;
            #pragma unroll
            for (int v = 0; v < 8; ++v)
                bfb.u[v] = *(const unsigned int*)&bt_lds[ncol * 36 + kmap(v, kb)];
            acc[a] = __builtin_amdgcn_wmma_f32_16x16x32_bf16(
                false, af.v, false, bfb.v, (short)0, acc[a], false, false);
        }
        __syncthreads();
    }

    #pragma unroll
    for (int a = 0; a < 4; ++a) {
        int ncol = ntile + a * 16 + m_lane;
        #pragma unroll
        for (int r = 0; r < 8; ++r) {
            int m = mtile + wave * 16 + r + half_off;
            out[(size_t)m * D_MODEL + ncol] = acc[a][r];
        }
    }
}

// ---------------------------------------------------------------------------
extern "C" void kernel_launch(void* const* d_in, const int* in_sizes, int n_in,
                              void* d_out, int out_size, void* d_ws, size_t ws_size,
                              hipStream_t stream) {
    const float* hidden   = (const float*)d_in[0];
    const float* Wq       = (const float*)d_in[1];
    const float* Wk       = (const float*)d_in[2];
    const float* Wv       = (const float*)d_in[3];
    const float* Wo       = (const float*)d_in[4];
    const float* rel_bias = (const float*)d_in[5];
    float* out = (float*)d_out;

    const size_t qkv_elems = (size_t)BATCH * H_NUM * S_LEN * DKV;   // 4M bf16
    const size_t att_elems = (size_t)BATCH * S_LEN * D_MODEL;       // 4M bf16

    char* ws = (char*)d_ws;
    unsigned short* qw  = (unsigned short*)ws;              ws += qkv_elems * 2;
    unsigned short* kw  = (unsigned short*)ws;              ws += qkv_elems * 2;
    unsigned short* vw  = (unsigned short*)ws;              ws += qkv_elems * 2;
    unsigned short* att = (unsigned short*)ws;              ws += att_elems * 2;
    float* table        = (float*)ws;

    bias_table_kernel<<<dim3(256), dim3(256), 0, stream>>>(rel_bias, table);
    qkv_gemm_kernel<<<dim3(D_MODEL / 64, (BATCH * S_LEN) / 64, 3), dim3(128), 0, stream>>>(
        hidden, Wq, Wk, Wv, qw, kw, vw);
    attn_kernel<<<dim3(S_LEN / 64, BATCH * H_NUM), dim3(128), 0, stream>>>(
        qw, kw, vw, table, att);
    out_gemm_kernel<<<dim3(D_MODEL / 64, (BATCH * S_LEN) / 64), dim3(128), 0, stream>>>(
        att, Wo, out);
}
